// EPTLayerAdaLNZero_79963701117018
// MI455X (gfx1250) — compile-verified, matrix operand order checked by reading, not run on hardware
//
#include <hip/hip_runtime.h>
#include <hip/hip_bf16.h>
#include <math.h>

// Problem constants (from reference): B,N,D,NH,DFFN,P = 2,1024,512,16,2048,4 ; DH=32
#define B_    2
#define N_    1024
#define D_    512
#define NH_   16
#define DFFN_ 2048
#define P_    4
#define DH_   32

typedef __attribute__((ext_vector_type(16))) __bf16 v16bf;
typedef __attribute__((ext_vector_type(8)))  float  v8f;

#define V8F_ZERO {0.f,0.f,0.f,0.f,0.f,0.f,0.f,0.f}

__device__ __forceinline__ unsigned short f2bf(float f) {
    unsigned int u = __float_as_uint(f);
    u = (u + 0x7FFFu + ((u >> 16) & 1u)) >> 16;   // round-to-nearest-even
    return (unsigned short)u;
}
__device__ __forceinline__ float bf2f(unsigned short h) {
    return __uint_as_float(((unsigned int)h) << 16);
}

union FragU { uint4 u[2]; v16bf v; };
// 32 contiguous bytes (16 bf16) -> one WMMA A/B fragment slice for this lane.
__device__ __forceinline__ v16bf load_frag(const unsigned short* p) {
    FragU f;
    f.u[0] = *(const uint4*)(p);
    f.u[1] = *(const uint4*)(p + 8);
    return f.v;
}
__device__ __forceinline__ v8f wmma_bf16(v16bf a, v16bf b, v8f c) {
    return __builtin_amdgcn_wmma_f32_16x16x32_bf16(false, a, false, b, (short)0, c, false, false);
}

// gfx1250 async global->LDS copy (ASYNCcnt-tracked). LDS byte offset is the
// low 32 bits of the generic pointer (flat LDS aperture: addr[31:0]).
__device__ __forceinline__ unsigned int lds_off(const void* p) {
    return (unsigned int)(unsigned long long)p;
}
__device__ __forceinline__ void async_copy_b128(unsigned int ldsaddr, const void* gaddr) {
    asm volatile("global_load_async_to_lds_b128 %0, %1, off"
                 :: "v"(ldsaddr), "v"(gaddr) : "memory");
}
__device__ __forceinline__ void wait_async_le1() {
    asm volatile("s_wait_asynccnt 0x1" ::: "memory");
}
__device__ __forceinline__ void wait_async_0() {
    asm volatile("s_wait_asynccnt 0x0" ::: "memory");
}

// ---------------------------------------------------------------------------
// K-cast: fp32 -> bf16
// ---------------------------------------------------------------------------
__global__ void k_cast_bf16(const float* __restrict__ src, unsigned short* __restrict__ dst, int n) {
    for (int i = blockIdx.x * blockDim.x + threadIdx.x; i < n; i += gridDim.x * blockDim.x)
        dst[i] = f2bf(src[i]);
}

// ---------------------------------------------------------------------------
// Conditioning: cond_mean = prompt.mean(P) ; silu ; tiny GEMMs for
// co_A/co_F (adaLN scale/shift) and alpha1/alpha2. One block, ~3 MFLOP.
// ---------------------------------------------------------------------------
__global__ __launch_bounds__(256)
void k_cond(const float* __restrict__ prompt,
            const float* __restrict__ W_adaA, const float* __restrict__ b_adaA,
            const float* __restrict__ W_adaF, const float* __restrict__ b_adaF,
            const float* __restrict__ Wa1,   const float* __restrict__ ba1,
            const float* __restrict__ Wa2,   const float* __restrict__ ba2,
            float* __restrict__ coA, float* __restrict__ coF,
            float* __restrict__ a1,  float* __restrict__ a2) {
    __shared__ float sm[B_ * D_];   // cond_mean
    __shared__ float ss[B_ * D_];   // silu(cond_mean)
    const int tid = threadIdx.x;
    for (int idx = tid; idx < B_ * D_; idx += blockDim.x) {
        int b = idx / D_, d = idx % D_;
        float s = 0.f;
        for (int p = 0; p < P_; ++p) s += prompt[(size_t)(b * P_ + p) * D_ + d];
        s *= (1.0f / P_);
        sm[idx] = s;
        ss[idx] = s / (1.0f + __expf(-s));
    }
    __syncthreads();
    for (int idx = tid; idx < 6144; idx += blockDim.x) {
        if (idx < 2048) {
            int b = idx / 1024, o = idx % 1024;
            float acc = b_adaA[o];
            const float* w = W_adaA + (size_t)o * D_;
            const float* x = ss + b * D_;
            for (int k = 0; k < D_; ++k) acc += x[k] * w[k];
            coA[idx] = acc;
        } else if (idx < 4096) {
            int j = idx - 2048, b = j / 1024, o = j % 1024;
            float acc = b_adaF[o];
            const float* w = W_adaF + (size_t)o * D_;
            const float* x = ss + b * D_;
            for (int k = 0; k < D_; ++k) acc += x[k] * w[k];
            coF[j] = acc;
        } else if (idx < 5120) {
            int j = idx - 4096, b = j / 512, o = j % 512;
            float acc = ba1[o];
            const float* w = Wa1 + (size_t)o * D_;
            const float* x = sm + b * D_;
            for (int k = 0; k < D_; ++k) acc += x[k] * w[k];
            a1[j] = acc;
        } else {
            int j = idx - 5120, b = j / 512, o = j % 512;
            float acc = ba2[o];
            const float* w = Wa2 + (size_t)o * D_;
            const float* x = sm + b * D_;
            for (int k = 0; k < D_; ++k) acc += x[k] * w[k];
            a2[j] = acc;
        }
    }
}

// ---------------------------------------------------------------------------
// AdaLN: layernorm(x) * (1+scale) + shift, output bf16. 1 block / row of 512.
// ---------------------------------------------------------------------------
__global__ __launch_bounds__(128)
void k_adaln(const float* __restrict__ X, const float* __restrict__ co,
             unsigned short* __restrict__ out) {
    const int row = blockIdx.x;
    const int b = row / N_;
    const int t = threadIdx.x;
    const float* xr = X + (size_t)row * D_;
    float v[4];
    #pragma unroll
    for (int j = 0; j < 4; ++j) v[j] = xr[t + j * 128];
    __shared__ float red[128];
    float s = v[0] + v[1] + v[2] + v[3];
    red[t] = s; __syncthreads();
    for (int off = 64; off; off >>= 1) { if (t < off) red[t] += red[t + off]; __syncthreads(); }
    const float mean = red[0] * (1.0f / D_);
    __syncthreads();
    float s2 = 0.f;
    #pragma unroll
    for (int j = 0; j < 4; ++j) { float d = v[j] - mean; s2 += d * d; }
    red[t] = s2; __syncthreads();
    for (int off = 64; off; off >>= 1) { if (t < off) red[t] += red[t + off]; __syncthreads(); }
    const float rstd = rsqrtf(red[0] * (1.0f / D_) + 1e-5f);
    #pragma unroll
    for (int j = 0; j < 4; ++j) {
        int col = t + j * 128;
        float sc = co[(size_t)b * 1024 + col];
        float sh = co[(size_t)b * 1024 + 512 + col];
        out[(size_t)row * D_ + col] = f2bf((v[j] - mean) * rstd * (1.0f + sc) + sh);
    }
}

// ---------------------------------------------------------------------------
// bf16 WMMA GEMM: out[M,Nout] = A[M,K] @ W[Nout,K]^T (+bias)
// Block: 256 thr = 8 waves; tile 32 rows x 128 cols; 2 accumulators / wave.
// K stages of 64; A tile (32x64 bf16 = 4KB) double-buffered in LDS, filled
// with global_load_async_to_lds_b128 (ASYNCcnt) one stage ahead. The
// steady-state loop is peeled so it is branch-free: issue async(s+1) ->
// s_wait_asynccnt 1 -> barrier -> 4 WMMA -> barrier.
// mode: 0 = f32 ; 1 = bf16 ; 2 = silu->bf16 ; 3 = resid + alpha*(acc+bias)
// ---------------------------------------------------------------------------
__global__ __launch_bounds__(256)
void k_gemm(const unsigned short* __restrict__ A, const unsigned short* __restrict__ W,
            const float* __restrict__ bias, int M, int Nout, int K, int mode,
            unsigned short* __restrict__ outb, float* __restrict__ outf,
            const float* __restrict__ resid, const float* __restrict__ alpha) {
    const int tid  = threadIdx.x;
    const int lane = tid & 31;
    const int wave = tid >> 5;
    const int m0   = blockIdx.x * 32;
    const int col0 = blockIdx.y * 128 + wave * 16;
    const int lrow = lane & 15;
    const int lhalf = (lane >> 4) << 4;

    // double-buffered A tile: 2 x (32 rows x 64 cols) bf16
    __shared__ __align__(16) unsigned short sA[2][32 * 64];

    // this thread's async-copy slice: 8 bf16 (16B) per stage
    const int li = tid << 3;                // 0..2047
    const int ar = li >> 6;                 // staging row 0..31
    const int ac = li & 63;                 // staging col
    const unsigned short* aSrc = A + (size_t)(m0 + ar) * K + ac;
    const unsigned int ldsDst0 = lds_off(&sA[0][li]);
    const unsigned int ldsDst1 = lds_off(&sA[1][li]);

    v8f acc0 = V8F_ZERO, acc1 = V8F_ZERO;
    const unsigned short* wrow = W + (size_t)(col0 + lrow) * K + lhalf;
    const unsigned short* aL0 = &sA[0][(lrow)      * 64 + lhalf];
    const unsigned short* aL1 = &sA[0][(16 + lrow) * 64 + lhalf];
    const int bufStride = 32 * 64;

    const int S = K >> 6;                   // K/64 stages (>= 8 for all calls)

    auto compute_stage = [&](int s) {
        const int k0 = s << 6;
        const int boff = (s & 1) * bufStride;
        v16bf b0 = load_frag(wrow + k0);
        v16bf b1 = load_frag(wrow + k0 + 32);
        __builtin_prefetch(wrow + k0 + 128, 0, 1);   // W stream, two stages ahead
        v16bf a00 = load_frag(aL0 + boff);
        v16bf a01 = load_frag(aL0 + boff + 32);
        v16bf a10 = load_frag(aL1 + boff);
        v16bf a11 = load_frag(aL1 + boff + 32);
        acc0 = wmma_bf16(a00, b0, acc0);
        acc0 = wmma_bf16(a01, b1, acc0);
        acc1 = wmma_bf16(a10, b0, acc1);
        acc1 = wmma_bf16(a11, b1, acc1);
    };

    async_copy_b128(ldsDst0, aSrc);         // preload stage 0
    // steady state: branch-free pipeline, one stage always in flight
    for (int s = 0; s < S - 1; ++s) {
        async_copy_b128(((s + 1) & 1) ? ldsDst1 : ldsDst0, aSrc + (s + 1) * 64);
        wait_async_le1();                   // stage s landed, s+1 in flight
        __syncthreads();                    // all waves' slices visible
        compute_stage(s);
        __syncthreads();                    // done reading buf[s&1]
    }
    // drain: last stage
    wait_async_0();
    __syncthreads();
    compute_stage(S - 1);

    const int col = col0 + lrow;
    const float bc = bias ? bias[col] : 0.0f;
    const int rb0 = m0 + ((lane >> 4) << 3);
    #pragma unroll
    for (int t = 0; t < 2; ++t) {
        const int rbase = rb0 + t * 16;
        #pragma unroll
        for (int r = 0; r < 8; ++r) {
            const int row = rbase + r;
            float v = (t ? acc1[r] : acc0[r]) + bc;
            const size_t idx = (size_t)row * Nout + col;
            if (mode == 0)      outf[idx] = v;
            else if (mode == 1) outb[idx] = f2bf(v);
            else if (mode == 2) outb[idx] = f2bf(v / (1.0f + __expf(-v)));
            else {
                const int b = row / N_;
                outf[idx] = resid[idx] + alpha[(size_t)b * Nout + col] * v;
            }
        }
    }
}

// ---------------------------------------------------------------------------
// Build V_attn^T : [B, NH, 128, N] bf16 (transposed so attention PV
// B-fragments are contiguous 32B loads).
// ---------------------------------------------------------------------------
__global__ void k_pack_vaT(const unsigned short* __restrict__ Hv,
                           const unsigned short* __restrict__ Vv,
                           unsigned short* __restrict__ VaT) {
    const size_t total = (size_t)B_ * NH_ * 128 * N_;
    for (size_t idx = (size_t)blockIdx.x * blockDim.x + threadIdx.x; idx < total;
         idx += (size_t)gridDim.x * blockDim.x) {
        const int n = (int)(idx % N_);
        const int d = (int)((idx / N_) % 128);
        const int h = (int)((idx / ((size_t)N_ * 128)) % NH_);
        const int b = (int)(idx / ((size_t)N_ * 128 * NH_));
        unsigned short v;
        if (d < DH_) {
            v = Hv[(size_t)(b * N_ + n) * D_ + h * DH_ + d];
        } else {
            const int c = (d - DH_) >> 5, dd = (d - DH_) & 31;
            v = Vv[((size_t)(b * N_ + n) * 3 + c) * D_ + h * DH_ + dd];
        }
        VaT[idx] = v;
    }
}

// ---------------------------------------------------------------------------
// Flash attention: one 16-query tile per wave, 4 waves/block.
// S = Q(16x128)K^T via 8 wmma ; online softmax (shfl_xor within wave32
// halves) ; P transposed C->A layout through a per-wave LDS tile ; O += P V
// via 8 wmma per 32-key block.
// ---------------------------------------------------------------------------
__global__ __launch_bounds__(128)
void k_attn(const unsigned short* __restrict__ Hq, const unsigned short* __restrict__ Hk,
            const unsigned short* __restrict__ VaT,
            const float* __restrict__ rbf, const float* __restrict__ Db,
            const int* __restrict__ Hmask,
            unsigned short* __restrict__ Hres, unsigned short* __restrict__ Vres) {
    const int bh = blockIdx.x;
    const int b = bh / NH_, h = bh % NH_;
    const int lane = threadIdx.x & 31;
    const int wave = threadIdx.x >> 5;
    const int q0 = (blockIdx.y * 4 + wave) * 16;
    const int lrow = lane & 15;
    const int lsel = lane >> 4;
    const int lhalf = lsel << 4;
    const float factor = 0.088388347648318447f;   // 0.5 / sqrt(DH)

    __shared__ __align__(64) unsigned short sP[4][16 * 32];
    unsigned short* myP = sP[wave];

    v16bf qa[4];
    {
        const unsigned short* qp =
            Hq + (size_t)(b * N_ + q0 + lrow) * (NH_ * 128) + h * 128 + lhalf;
        #pragma unroll
        for (int f = 0; f < 4; ++f) qa[f] = load_frag(qp + f * 32);
    }

    v8f O[8];
    #pragma unroll
    for (int c8 = 0; c8 < 8; ++c8) O[c8] = (v8f)V8F_ZERO;
    float mv[8], lv[8];
    #pragma unroll
    for (int r = 0; r < 8; ++r) { mv[r] = -1e30f; lv[r] = 0.f; }

    for (int j0 = 0; j0 < N_; j0 += 32) {
        v8f s0 = V8F_ZERO, s1 = V8F_ZERO;
        const unsigned short* kp0 =
            Hk + (size_t)(b * N_ + j0 + lrow) * (NH_ * 128) + h * 128 + lhalf;
        const unsigned short* kp1 = kp0 + (size_t)16 * (NH_ * 128);
        #pragma unroll
        for (int f = 0; f < 4; ++f) {
            s0 = wmma_bf16(qa[f], load_frag(kp0 + f * 32), s0);
            s1 = wmma_bf16(qa[f], load_frag(kp1 + f * 32), s1);
        }
        const int ja = j0 + lrow, jb = ja + 16;
        const float mA = Hmask[b * N_ + ja] ? 0.f : -1e30f;
        const float mB = Hmask[b * N_ + jb] ? 0.f : -1e30f;

        float scl[8];
        #pragma unroll
        for (int r = 0; r < 8; ++r) {
            const int q = q0 + r + (lsel << 3);
            const size_t rb = ((size_t)(b * NH_ + h) * N_ + q) * N_;
            const size_t db = ((size_t)b * N_ + q) * N_;
            const float x0 = s0[r] * factor + rbf[rb + ja] + Db[db + ja] + mA;
            const float x1 = s1[r] * factor + rbf[rb + jb] + Db[db + jb] + mB;
            float mx = fmaxf(x0, x1);
            #pragma unroll
            for (int off = 8; off; off >>= 1) mx = fmaxf(mx, __shfl_xor(mx, off, 16));
            const float mnew = fmaxf(mv[r], mx);
            const float sc = __expf(mv[r] - mnew);
            mv[r] = mnew;
            const float p0 = __expf(x0 - mnew);
            const float p1 = __expf(x1 - mnew);
            float rs = p0 + p1;
            #pragma unroll
            for (int off = 8; off; off >>= 1) rs += __shfl_xor(rs, off, 16);
            lv[r] = lv[r] * sc + rs;
            scl[r] = sc;
            const int prow = r + (lsel << 3);
            myP[prow * 32 + lrow]      = f2bf(p0);
            myP[prow * 32 + 16 + lrow] = f2bf(p1);
        }
        #pragma unroll
        for (int c8 = 0; c8 < 8; ++c8)
            #pragma unroll
            for (int r = 0; r < 8; ++r) O[c8][r] *= scl[r];
        __syncthreads();
        const v16bf pa = load_frag(&myP[lrow * 32 + lhalf]);   // transposed P
        #pragma unroll
        for (int c8 = 0; c8 < 8; ++c8) {
            const unsigned short* vp =
                VaT + ((size_t)((b * NH_ + h) * 128 + c8 * 16 + lrow)) * N_ + j0 + lhalf;
            O[c8] = wmma_bf16(pa, load_frag(vp), O[c8]);
        }
        __syncthreads();
    }

    float inv[8];
    #pragma unroll
    for (int r = 0; r < 8; ++r) inv[r] = 1.0f / lv[r];
    #pragma unroll
    for (int c8 = 0; c8 < 8; ++c8) {
        const int d = c8 * 16 + lrow;
        #pragma unroll
        for (int r = 0; r < 8; ++r) {
            const int q = q0 + r + (lsel << 3);
            const float v = O[c8][r] * inv[r];
            if (d < DH_) {
                Hres[(size_t)(b * N_ + q) * D_ + h * DH_ + d] = f2bf(v);
            } else {
                const int c = (d - DH_) >> 5, dd = (d - DH_) & 31;
                Vres[((size_t)(b * N_ + q) * 3 + c) * D_ + h * DH_ + dd] = f2bf(v);
            }
        }
    }
}

// ---------------------------------------------------------------------------
// scaler = concat(H_norm2, ||V1||_c) as bf16 [B*N, 2D]
// ---------------------------------------------------------------------------
__global__ void k_scaler(const unsigned short* __restrict__ Hn2,
                         const float* __restrict__ Vproj,
                         unsigned short* __restrict__ scal) {
    const size_t total = (size_t)B_ * N_ * D_;
    for (size_t idx = (size_t)blockIdx.x * blockDim.x + threadIdx.x; idx < total;
         idx += (size_t)gridDim.x * blockDim.x) {
        const size_t row = idx / D_;
        const int d = (int)(idx % D_);
        scal[row * 1024 + d] = Hn2[row * D_ + d];
        float s = 0.f;
        #pragma unroll
        for (int c = 0; c < 3; ++c) {
            float v = Vproj[(row * 3 + c) * 1024 + d];
            s += v * v;
        }
        scal[row * 1024 + 512 + d] = f2bf(sqrtf(s));
    }
}

// ---------------------------------------------------------------------------
// Final: H_out = H2 + alpha2*H_ffn ; V_out = V_update (x) V2
// ---------------------------------------------------------------------------
__global__ void k_final(const float* __restrict__ H2, const float* __restrict__ a2,
                        const float* __restrict__ sout, const float* __restrict__ Vproj,
                        float* __restrict__ out) {
    const size_t total = (size_t)B_ * N_ * D_;
    float* Hout = out;
    float* Vout = out + total;
    for (size_t idx = (size_t)blockIdx.x * blockDim.x + threadIdx.x; idx < total;
         idx += (size_t)gridDim.x * blockDim.x) {
        const size_t row = idx / D_;
        const int d = (int)(idx % D_);
        const int b = (int)(row / N_);
        Hout[idx] = H2[idx] + a2[(size_t)b * D_ + d] * sout[row * 1024 + d];
        const float vu = sout[row * 1024 + 512 + d];
        #pragma unroll
        for (int c = 0; c < 3; ++c)
            Vout[(row * 3 + c) * D_ + d] = vu * Vproj[(row * 3 + c) * 1024 + 512 + d];
    }
}

// ---------------------------------------------------------------------------
extern "C" void kernel_launch(void* const* d_in, const int* in_sizes, int n_in,
                              void* d_out, int out_size, void* d_ws, size_t ws_size,
                              hipStream_t stream) {
    (void)in_sizes; (void)n_in; (void)out_size; (void)ws_size;
    const float* H     = (const float*)d_in[0];
    const float* V     = (const float*)d_in[1];
    const float* Db    = (const float*)d_in[2];
    const float* rbf   = (const float*)d_in[3];
    const int*   Hmask = (const int*)d_in[4];
    const float* prompt= (const float*)d_in[5];
    const float* W_adaA= (const float*)d_in[6];  const float* b_adaA=(const float*)d_in[7];
    const float* Wq    = (const float*)d_in[8];  const float* bq    =(const float*)d_in[9];
    const float* Wk    = (const float*)d_in[10]; const float* bk    =(const float*)d_in[11];
    const float* Wv    = (const float*)d_in[12]; const float* bv    =(const float*)d_in[13];
    const float* Wvv   = (const float*)d_in[14];
    const float* Wo    = (const float*)d_in[15]; const float* bo    =(const float*)d_in[16];
    const float* Wvo   = (const float*)d_in[17];
    const float* Wa1   = (const float*)d_in[18]; const float* ba1   =(const float*)d_in[19];
    const float* W_adaF= (const float*)d_in[20]; const float* b_adaF=(const float*)d_in[21];
    const float* Wlv   = (const float*)d_in[22];
    const float* W1    = (const float*)d_in[23]; const float* b1    =(const float*)d_in[24];
    const float* W2    = (const float*)d_in[25]; const float* b2    =(const float*)d_in[26];
    const float* Wa2   = (const float*)d_in[27]; const float* ba2   =(const float*)d_in[28];

    // ---- workspace carving (~125 MB total) ----
    char* p = (char*)d_ws;
    auto carve = [&](size_t bytes) -> void* {
        void* r = (void*)p;
        p += (bytes + 255) & ~(size_t)255;
        return r;
    };
    const size_t szH  = (size_t)B_ * N_ * D_;     // 1M
    const size_t szV3 = szH * 3;                  // 3M
    const size_t szQ  = (size_t)B_ * N_ * 4 * D_; // 4M

    float* coA   = (float*)carve(2048 * 4);
    float* coF   = (float*)carve(2048 * 4);
    float* a1    = (float*)carve(1024 * 4);
    float* a2    = (float*)carve(1024 * 4);
    float* H2    = (float*)carve(szH * 4);
    float* Vproj = (float*)carve((size_t)B_ * N_ * 3 * 1024 * 4);
    float* sout  = (float*)carve((size_t)B_ * N_ * 1024 * 4);

    unsigned short* Hn1  = (unsigned short*)carve(szH * 2);
    unsigned short* Vb   = (unsigned short*)carve(szV3 * 2);
    unsigned short* Wq_b = (unsigned short*)carve((size_t)2048 * 512 * 2);
    unsigned short* Wk_b = (unsigned short*)carve((size_t)2048 * 512 * 2);
    unsigned short* Wv_b = (unsigned short*)carve((size_t)512 * 512 * 2);
    unsigned short* Wvv_b= (unsigned short*)carve((size_t)512 * 512 * 2);
    unsigned short* Wo_b = (unsigned short*)carve((size_t)512 * 512 * 2);
    unsigned short* Wvo_b= (unsigned short*)carve((size_t)512 * 512 * 2);
    unsigned short* Wlv_b= (unsigned short*)carve((size_t)1024 * 512 * 2);
    unsigned short* W1_b = (unsigned short*)carve((size_t)2048 * 1024 * 2);
    unsigned short* W2_b = (unsigned short*)carve((size_t)1024 * 2048 * 2);
    unsigned short* Hq_b = (unsigned short*)carve(szQ * 2);
    unsigned short* Hk_b = (unsigned short*)carve(szQ * 2);
    unsigned short* Hv_t = (unsigned short*)carve(szH * 2);
    unsigned short* Vv_t = (unsigned short*)carve(szV3 * 2);
    unsigned short* VaT  = (unsigned short*)carve(szQ * 2);
    unsigned short* HresB= (unsigned short*)carve(szH * 2);
    unsigned short* VresB= (unsigned short*)carve(szV3 * 2);
    unsigned short* VnewB= (unsigned short*)carve(szV3 * 2);
    unsigned short* Hn2  = (unsigned short*)carve(szH * 2);
    unsigned short* scal = (unsigned short*)carve((size_t)B_ * N_ * 1024 * 2);
    unsigned short* midB = (unsigned short*)carve((size_t)B_ * N_ * DFFN_ * 2);

    auto cast = [&](const float* s, unsigned short* d, int n) {
        int blocks = (n + 255) / 256; if (blocks > 4096) blocks = 4096;
        hipLaunchKernelGGL(k_cast_bf16, dim3(blocks), dim3(256), 0, stream, s, d, n);
    };

    // 1) weight + V casts (weights ~16 MB bf16: L2-resident afterwards)
    cast(Wq,  Wq_b,  2048 * 512);
    cast(Wk,  Wk_b,  2048 * 512);
    cast(Wv,  Wv_b,   512 * 512);
    cast(Wvv, Wvv_b,  512 * 512);
    cast(Wo,  Wo_b,   512 * 512);
    cast(Wvo, Wvo_b,  512 * 512);
    cast(Wlv, Wlv_b, 1024 * 512);
    cast(W1,  W1_b,  2048 * 1024);
    cast(W2,  W2_b,  1024 * 2048);
    cast(V,   Vb,    (int)szV3);

    // 2) conditioning
    hipLaunchKernelGGL(k_cond, dim3(1), dim3(256), 0, stream,
                       prompt, W_adaA, b_adaA, W_adaF, b_adaF, Wa1, ba1, Wa2, ba2,
                       coA, coF, a1, a2);

    // 3) adaln #1 -> H_norm bf16
    hipLaunchKernelGGL(k_adaln, dim3(B_ * N_), dim3(128), 0, stream, H, coA, Hn1);

    // 4) projections (all v_wmma_f32_16x16x32_bf16; M tiles of 32 rows)
    hipLaunchKernelGGL(k_gemm, dim3(64, 16), dim3(256), 0, stream,
                       Hn1, Wq_b, bq, 2048, 2048, 512, 1, Hq_b, (float*)0, (const float*)0, (const float*)0);
    hipLaunchKernelGGL(k_gemm, dim3(64, 16), dim3(256), 0, stream,
                       Hn1, Wk_b, bk, 2048, 2048, 512, 1, Hk_b, (float*)0, (const float*)0, (const float*)0);
    hipLaunchKernelGGL(k_gemm, dim3(64, 4), dim3(256), 0, stream,
                       Hn1, Wv_b, bv, 2048, 512, 512, 1, Hv_t, (float*)0, (const float*)0, (const float*)0);
    hipLaunchKernelGGL(k_gemm, dim3(192, 4), dim3(256), 0, stream,
                       Vb, Wvv_b, (const float*)0, 6144, 512, 512, 1, Vv_t, (float*)0, (const float*)0, (const float*)0);

    // 5) pack V_attn^T
    hipLaunchKernelGGL(k_pack_vaT, dim3(4096), dim3(256), 0, stream, Hv_t, Vv_t, VaT);

    // 6) flash attention
    hipLaunchKernelGGL(k_attn, dim3(B_ * NH_, N_ / 64), dim3(128), 0, stream,
                       Hq_b, Hk_b, VaT, rbf, Db, Hmask, HresB, VresB);

    // 7) output projections; Wo fused with residual: H2 = H + alpha1*(HresB@Wo^T+bo)
    hipLaunchKernelGGL(k_gemm, dim3(64, 4), dim3(256), 0, stream,
                       HresB, Wo_b, bo, 2048, 512, 512, 3, (unsigned short*)0, H2, H, a1);
    hipLaunchKernelGGL(k_gemm, dim3(192, 4), dim3(256), 0, stream,
                       VresB, Wvo_b, (const float*)0, 6144, 512, 512, 1, VnewB, (float*)0, (const float*)0, (const float*)0);

    // 8) adaln #2 on H2
    hipLaunchKernelGGL(k_adaln, dim3(B_ * N_), dim3(128), 0, stream, H2, coF, Hn2);

    // 9) V_proj = V_new @ Wlv^T  (fp32 out, needed for V1 norm and V2)
    hipLaunchKernelGGL(k_gemm, dim3(192, 8), dim3(256), 0, stream,
                       VnewB, Wlv_b, (const float*)0, 6144, 1024, 512, 0, (unsigned short*)0, Vproj, (const float*)0, (const float*)0);

    // 10) scaler = [H_norm2 | ||V1||]
    hipLaunchKernelGGL(k_scaler, dim3(2048), dim3(256), 0, stream, Hn2, Vproj, scal);

    // 11) FFN: silu(scaler@W1^T+b1) @ W2^T + b2
    hipLaunchKernelGGL(k_gemm, dim3(64, 16), dim3(256), 0, stream,
                       scal, W1_b, b1, 2048, 2048, 1024, 2, midB, (float*)0, (const float*)0, (const float*)0);
    hipLaunchKernelGGL(k_gemm, dim3(64, 8), dim3(256), 0, stream,
                       midB, W2_b, b2, 2048, 1024, 2048, 0, (unsigned short*)0, sout, (const float*)0, (const float*)0);

    // 12) residual #2 + V outer product -> d_out
    hipLaunchKernelGGL(k_final, dim3(2048), dim3(256), 0, stream,
                       H2, a2, sout, Vproj, (float*)d_out);
}
